// AttnPointNetConv_18227841204607
// MI455X (gfx1250) — compile-verified, matrix operand order checked by reading, not run on hardware
//
#include <hip/hip_runtime.h>
#include <math.h>

typedef __attribute__((ext_vector_type(16))) _Float16 v16h;
typedef __attribute__((ext_vector_type(8)))  float    v8f;

#define IN_CH 61
#define DK 64      // d_in = 61 + 3
#define DN 64      // out channels
#define FEAT_STRIDE 72  // padded LDS row stride (halves), conflict-free

// ---------------------------------------------------------------------------
// Kernel 0: init out / per-node max / per-node sum / stats accumulators
// ---------------------------------------------------------------------------
__global__ void k_init(float* __restrict__ out, float* __restrict__ M,
                       float* __restrict__ S, float* STATS, int nNodes) {
  int idx = blockIdx.x * blockDim.x + threadIdx.x;
  int total = nNodes * DN;
  if (idx < total) out[idx] = 0.0f;
  if (idx < nNodes) { M[idx] = -__builtin_inff(); S[idx] = 0.0f; }
  if (idx < 260) STATS[idx] = 0.0f;   // chSum[64] chSq[64] gStat[2] scale/shift[128] gss[2]
}

// ---------------------------------------------------------------------------
// Kernel 1: per-edge feature build + 16x64 @ 64x64 GEMM via WMMA + BN stats
// Block = 128 threads = 4 waves; each wave owns a 16-edge tile per iteration.
// ---------------------------------------------------------------------------
__global__ __launch_bounds__(128) void k_edge_gemm(
    const float* __restrict__ x, const float* __restrict__ pos,
    const int* __restrict__ ei, const float* __restrict__ W1,
    const float* __restrict__ b1, float* __restrict__ H,
    float* __restrict__ chSum, float* __restrict__ chSq,
    int nEdges, int nTiles) {
  __shared__ _Float16 Wlds[DK * DN];                 // [k][n], f16
  __shared__ float    bLds[DN];
  __shared__ _Float16 featLds[4][16 * FEAT_STRIDE];  // per-wave 16 x 64 (padded)

  const int tid = threadIdx.x;
  for (int i = tid; i < DK * DN; i += 128) Wlds[i] = (_Float16)W1[i];
  if (tid < DN) bLds[tid] = b1[tid];
  __syncthreads();

  const int wave   = tid >> 5;
  const int lane   = tid & 31;
  const int laneM  = lane & 15;
  const int laneHi = lane >> 4;   // 0: lanes 0-15, 1: lanes 16-31

  // B fragments (kept resident): 4 N-tiles x 2 K-chunks.
  // 16-bit fragment layout: lanes 0-15 hold K {0..7,16..23}(+32*kc) for col n,
  // lanes 16-31 hold K {8..15,24..31}(+32*kc).
  v16h bf[4][2];
#pragma unroll
  for (int t = 0; t < 4; ++t) {
    const int n = t * 16 + laneM;
#pragma unroll
    for (int kc = 0; kc < 2; ++kc) {
      const int K0 = kc * 32 + (laneHi ? 8 : 0);
      const int K1 = kc * 32 + (laneHi ? 24 : 16);
      v16h bb;
#pragma unroll
      for (int v = 0; v < 4; ++v) {
        bb[2 * v]     = Wlds[(K0 + 2 * v) * DN + n];
        bb[2 * v + 1] = Wlds[(K0 + 2 * v + 1) * DN + n];
        bb[8 + 2 * v] = Wlds[(K1 + 2 * v) * DN + n];
        bb[9 + 2 * v] = Wlds[(K1 + 2 * v + 1) * DN + n];
      }
      bf[t][kc] = bb;
    }
  }

  float ssum[4] = {0.f, 0.f, 0.f, 0.f};
  float ssq[4]  = {0.f, 0.f, 0.f, 0.f};

  _Float16* frow = &featLds[wave][laneM * FEAT_STRIDE];
  const int mbase = laneHi * 8;

  const int waveGlobal = blockIdx.x * 4 + wave;
  const int waveStride = gridDim.x * 4;
  for (int tile = waveGlobal; tile < nTiles; tile += waveStride) {
    const int e  = tile * 16 + laneM;
    const int ec = (e < nEdges) ? e : (nEdges - 1);
    const int src = ei[ec];
    const int dst = ei[nEdges + ec];
    const float* xr = x + (size_t)src * IN_CH;
    const float* ps = pos + (size_t)src * 3;
    const float* pd = pos + (size_t)dst * 3;
    // Stage feat row (this lane covers halves [laneHi*32, laneHi*32+32))
    const int kbeg = laneHi * 32;
#pragma unroll 8
    for (int k = kbeg; k < kbeg + 32; ++k) {
      float v = (k < IN_CH) ? xr[k] : (ps[k - IN_CH] - pd[k - IN_CH]);
      frow[k] = (_Float16)v;
    }
    // cross-lane LDS RAW: wait for all DS stores of this wave
    asm volatile("s_wait_dscnt 0x0" ::: "memory");

    // A fragments for the two K-chunks
    v16h a[2];
#pragma unroll
    for (int kc = 0; kc < 2; ++kc) {
      const int K0 = kc * 32 + (laneHi ? 8 : 0);
      const int K1 = kc * 32 + (laneHi ? 24 : 16);
      v16h av;
#pragma unroll
      for (int v = 0; v < 4; ++v) {
        av[2 * v]     = frow[K0 + 2 * v];
        av[2 * v + 1] = frow[K0 + 2 * v + 1];
        av[8 + 2 * v] = frow[K1 + 2 * v];
        av[9 + 2 * v] = frow[K1 + 2 * v + 1];
      }
      a[kc] = av;
    }

    // Per-lane store base: row (tile*16 + mbase), column laneM (+ t*16 imm)
    float* Hp = H + ((size_t)(tile * 16 + mbase)) * DN + laneM;
    const bool fullTile = (tile * 16 + 16) <= nEdges;

#pragma unroll
    for (int t = 0; t < 4; ++t) {
      const float bv = bLds[t * 16 + laneM];
      v8f c;
#pragma unroll
      for (int r = 0; r < 8; ++r) c[r] = bv;
      c = __builtin_amdgcn_wmma_f32_16x16x32_f16(false, a[0], false, bf[t][0],
                                                 (short)0, c, false, false);
      c = __builtin_amdgcn_wmma_f32_16x16x32_f16(false, a[1], false, bf[t][1],
                                                 (short)0, c, false, false);
      if (fullTile) {
        // fast path: unguarded, base + immediate-offset stores
#pragma unroll
        for (int r = 0; r < 8; ++r) {
          const float vv = c[r];
          Hp[(size_t)r * DN + t * 16] = vv;
          ssum[t] += vv;
          ssq[t]  += vv * vv;
        }
      } else {
#pragma unroll
        for (int r = 0; r < 8; ++r) {
          const int row = tile * 16 + mbase + r;
          const float vv = c[r];
          if (row < nEdges) {
            Hp[(size_t)r * DN + t * 16] = vv;
            ssum[t] += vv;
            ssq[t]  += vv * vv;
          }
        }
      }
    }
  }

#pragma unroll
  for (int t = 0; t < 4; ++t) {
    atomicAdd(&chSum[t * 16 + laneM], ssum[t]);
    atomicAdd(&chSq[t * 16 + laneM], ssq[t]);
  }
}

// ---------------------------------------------------------------------------
// Kernel 2: finalize BN1 scale/shift (64 channels)
// ---------------------------------------------------------------------------
__global__ void k_finalize_bn(float* STATS, const float* __restrict__ g1,
                              const float* __restrict__ be1, float invE) {
  int c = threadIdx.x;
  float mu  = STATS[c] * invE;
  float var = STATS[64 + c] * invE - mu * mu;
  float sc  = g1[c] * rsqrtf(var + 1e-5f);
  STATS[130 + c] = sc;
  STATS[194 + c] = be1[c] - mu * sc;
}

// ---------------------------------------------------------------------------
// Kernel 3: h = silu(bn(hpre)) in-place; gpre = h.Wg + bg; gate stats
// ---------------------------------------------------------------------------
__global__ __launch_bounds__(256) void k_bnsilu_gate(
    const float* STATS, const float* __restrict__ Wg, const float* __restrict__ bg,
    float* __restrict__ H, float* __restrict__ G, float* gStat, int nEdges) {
  __shared__ float sSc[DN], sSh[DN], sWg[DN];
  const int tid = threadIdx.x;
  if (tid < DN) { sSc[tid] = STATS[130 + tid]; sSh[tid] = STATS[194 + tid]; sWg[tid] = Wg[tid]; }
  __syncthreads();
  const float bgv = bg[0];
  float lsum = 0.f, lsq = 0.f;
  for (int e = blockIdx.x * blockDim.x + tid; e < nEdges; e += gridDim.x * blockDim.x) {
    float* row = H + (size_t)e * DN;
    float acc = bgv;
#pragma unroll
    for (int q = 0; q < 16; ++q) {
      float4 v4 = *(float4*)(row + q * 4);
      float a0 = v4.x * sSc[4 * q + 0] + sSh[4 * q + 0];
      float a1 = v4.y * sSc[4 * q + 1] + sSh[4 * q + 1];
      float a2 = v4.z * sSc[4 * q + 2] + sSh[4 * q + 2];
      float a3 = v4.w * sSc[4 * q + 3] + sSh[4 * q + 3];
      a0 = a0 * (1.f / (1.f + __expf(-a0)));
      a1 = a1 * (1.f / (1.f + __expf(-a1)));
      a2 = a2 * (1.f / (1.f + __expf(-a2)));
      a3 = a3 * (1.f / (1.f + __expf(-a3)));
      v4.x = a0; v4.y = a1; v4.z = a2; v4.w = a3;
      *(float4*)(row + q * 4) = v4;
      acc += a0 * sWg[4 * q] + a1 * sWg[4 * q + 1] + a2 * sWg[4 * q + 2] + a3 * sWg[4 * q + 3];
    }
    G[e] = acc;
    lsum += acc; lsq += acc * acc;
  }
  // wave32 reduction then one atomic per wave
  for (int off = 16; off > 0; off >>= 1) {
    lsum += __shfl_xor(lsum, off, 32);
    lsq  += __shfl_xor(lsq,  off, 32);
  }
  if ((tid & 31) == 0) { atomicAdd(&gStat[0], lsum); atomicAdd(&gStat[1], lsq); }
}

// ---------------------------------------------------------------------------
// Kernel 4: finalize gate BN (1 channel)
// ---------------------------------------------------------------------------
__global__ void k_finalize_gate(float* STATS, const float* __restrict__ gg,
                                const float* __restrict__ bgb, float invE) {
  float mu  = STATS[128] * invE;
  float var = STATS[129] * invE - mu * mu;
  float sc  = gg[0] * rsqrtf(var + 1e-5f);
  STATS[258] = sc;
  STATS[259] = bgb[0] - mu * sc;
}

__device__ inline void atomicMaxFloat(float* addr, float val) {
  if (val >= 0.0f) atomicMax((int*)addr, __float_as_int(val));
  else             atomicMin((unsigned int*)addr, __float_as_uint(val));
}

// ---------------------------------------------------------------------------
// Kernel 5: gate = silu(bn(gpre)) in-place; segment max over dst
// ---------------------------------------------------------------------------
__global__ void k_gate(const float* STATS, float* __restrict__ G,
                       const int* __restrict__ ei, float* __restrict__ M, int nEdges) {
  int e = blockIdx.x * blockDim.x + threadIdx.x;
  if (e >= nEdges) return;
  float v = G[e] * STATS[258] + STATS[259];
  v = v * (1.f / (1.f + __expf(-v)));
  G[e] = v;
  atomicMaxFloat(&M[ei[nEdges + e]], v);
}

// ---------------------------------------------------------------------------
// Kernel 6: e = exp(gate - m[dst]) in-place; segment sum
// ---------------------------------------------------------------------------
__global__ void k_expsum(float* __restrict__ G, const int* __restrict__ ei,
                         const float* __restrict__ M, float* __restrict__ S, int nEdges) {
  int e = blockIdx.x * blockDim.x + threadIdx.x;
  if (e >= nEdges) return;
  int dst = ei[nEdges + e];
  float ev = __expf(G[e] - M[dst]);
  G[e] = ev;
  atomicAdd(&S[dst], ev);
}

// ---------------------------------------------------------------------------
// Kernel 7: out[dst] += alpha * h[e]   (one thread = one edge x 4 channels)
// ---------------------------------------------------------------------------
__global__ void k_scatter(const float* __restrict__ H, const float* __restrict__ G,
                          const float* __restrict__ S, const int* __restrict__ ei,
                          float* __restrict__ out, int nEdges) {
  int idx = blockIdx.x * blockDim.x + threadIdx.x;
  int total = nEdges * 16;
  if (idx >= total) return;
  int e = idx >> 4;
  int q = idx & 15;
  int dst = ei[nEdges + e];
  float alpha = G[e] / (S[dst] + 1e-16f);
  const float4 hv = *(const float4*)(H + (size_t)e * DN + q * 4);
  float* o = out + (size_t)dst * DN + q * 4;
  atomicAdd(o + 0, alpha * hv.x);
  atomicAdd(o + 1, alpha * hv.y);
  atomicAdd(o + 2, alpha * hv.z);
  atomicAdd(o + 3, alpha * hv.w);
}

// ---------------------------------------------------------------------------
extern "C" void kernel_launch(void* const* d_in, const int* in_sizes, int n_in,
                              void* d_out, int out_size, void* d_ws, size_t ws_size,
                              hipStream_t stream) {
  const float* x   = (const float*)d_in[0];
  const float* pos = (const float*)d_in[1];
  const int*   ei  = (const int*)d_in[2];
  const float* W1  = (const float*)d_in[3];
  const float* b1  = (const float*)d_in[4];
  const float* g1  = (const float*)d_in[5];
  const float* be1 = (const float*)d_in[6];
  const float* Wg  = (const float*)d_in[7];
  const float* bg  = (const float*)d_in[8];
  const float* gg  = (const float*)d_in[9];
  const float* bgb = (const float*)d_in[10];

  const int nNodes = in_sizes[0] / IN_CH;
  const int nEdges = in_sizes[2] / 2;
  float* out = (float*)d_out;

  // workspace layout (floats)
  float* ws    = (float*)d_ws;
  float* H     = ws;                               // [E,64] hpre -> h (in place)
  float* G     = H + (size_t)nEdges * DN;          // [E] gpre -> gate -> exp
  float* Mbuf  = G + nEdges;                       // [N] segment max
  float* Sbuf  = Mbuf + nNodes;                    // [N] segment sum
  float* STATS = Sbuf + nNodes;                    // [260]

  const float invE = 1.0f / (float)nEdges;

  {
    int total = nNodes * DN;
    k_init<<<(total + 255) / 256, 256, 0, stream>>>(out, Mbuf, Sbuf, STATS, nNodes);
  }
  {
    int nTiles = (nEdges + 15) / 16;
    k_edge_gemm<<<2048, 128, 0, stream>>>(x, pos, ei, W1, b1, H,
                                          &STATS[0], &STATS[64], nEdges, nTiles);
  }
  k_finalize_bn<<<1, 64, 0, stream>>>(STATS, g1, be1, invE);
  k_bnsilu_gate<<<2048, 256, 0, stream>>>(STATS, Wg, bg, H, G, &STATS[128], nEdges);
  k_finalize_gate<<<1, 1, 0, stream>>>(STATS, gg, bgb, invE);

  int eb = (nEdges + 255) / 256;
  k_gate<<<eb, 256, 0, stream>>>(STATS, G, ei, Mbuf, nEdges);
  k_expsum<<<eb, 256, 0, stream>>>(G, ei, Mbuf, Sbuf, nEdges);

  int total16 = nEdges * 16;
  k_scatter<<<(total16 + 255) / 256, 256, 0, stream>>>(H, G, Sbuf, ei, out, nEdges);
}
